// HyenaOperator_40827959116617
// MI455X (gfx1250) — matrix operand order repaired and checked by reading, not compiled, not run on hardware
//
#include <hip/hip_runtime.h>
#include <hip/hip_bf16.h>
#include <math.h>

#define BATCH   4
#define DMODEL  1024
#define LSEQ    2048
#define INNER3  3072
#define NFILT   64
#define KREV_ROW 2112   // 2048 filter taps (reversed) + 64 zero tail
#define VX_ROW   2112   // 64 zero head + 2048 sequence
#define VX_PAD   64

typedef __bf16 bf16_t;
typedef bf16_t v16bf __attribute__((ext_vector_type(16)));
typedef bf16_t v8bf  __attribute__((ext_vector_type(8)));
typedef float  v8f   __attribute__((ext_vector_type(8)));

static __device__ __forceinline__ v8f wmma_bf16(v16bf a, v16bf b, v8f c) {
  // D = A(16x32 bf16) * B(32x16 bf16) + C(16x16 f32)
  return __builtin_amdgcn_wmma_f32_16x16x32_bf16(false, a, false, b, (short)0, c,
                                                 false, false);
}

// Load 16 bf16 as two aligned 16B chunks -> one v16bf fragment
static __device__ __forceinline__ v16bf ld_pair(const bf16_t* p0, const bf16_t* p1) {
  v8bf lo = *(const v8bf*)p0;
  v8bf hi = *(const v8bf*)p1;
  return __builtin_shufflevector(lo, hi, 0,1,2,3,4,5,6,7,8,9,10,11,12,13,14,15);
}

// ---------------------------------------------------------------- utilities
__global__ __launch_bounds__(256) void k_f32_to_bf16(const float* __restrict__ s,
                                                     bf16_t* __restrict__ d, int n) {
  int i = blockIdx.x * 256 + threadIdx.x;
  if (i < n) d[i] = (bf16_t)s[i];
}

__global__ __launch_bounds__(256) void k_transpose_bf16(const float* __restrict__ w,
                                                        bf16_t* __restrict__ wt,
                                                        int K, int N) {
  int i = blockIdx.x * 256 + threadIdx.x;
  if (i >= K * N) return;
  int n = i / K, k = i % K;
  wt[(size_t)n * K + k] = (bf16_t)w[(size_t)k * N + n];  // wt[n][k] = w[k][n]
}

__global__ __launch_bounds__(256) void k_zero_bf16(bf16_t* __restrict__ p, int n) {
  int i = blockIdx.x * 256 + threadIdx.x;
  if (i < n) p[i] = (bf16_t)0.0f;
}

// ------------------------------------------------- filter generator (sin-MLP)
// Writes krev[c][2047-l] = k[c][l] in bf16 (reversed so Toeplitz A-loads are
// forward-contiguous); tail [2048,2112) stays zero (k_zero_bf16 ran first).
__global__ __launch_bounds__(64) void k_filter(
    const float* __restrict__ w1, const float* __restrict__ b1,
    const float* __restrict__ w2, const float* __restrict__ b2,
    const float* __restrict__ w3, const float* __restrict__ b3,
    const float* __restrict__ wo, const float* __restrict__ freq,
    bf16_t* __restrict__ krev) {
  int l = blockIdx.x;
  int f = threadIdx.x;
  __shared__ float hA[NFILT], hB[NFILT];
  float t  = (float)l * (1.0f / 2047.0f);
  float wa = 6.28318530717958647692f * (float)l * (1.0f / 2048.0f);
  float fw = 1.0e-4f * wa;
  float z0 = t, z1 = cosf(fw), z2 = -sinf(fw);
  float fr = freq[f];
  float s = z0 * w1[f] + z1 * w1[NFILT + f] + z2 * w1[2 * NFILT + f] + b1[f];
  hA[f] = sinf(fr * s);
  __syncthreads();
  float acc = b2[f];
  for (int q = 0; q < NFILT; q++) acc += hA[q] * w2[q * NFILT + f];
  float h2 = sinf(fr * acc);
  __syncthreads();
  hB[f] = h2;
  __syncthreads();
  acc = b3[f];
  for (int q = 0; q < NFILT; q++) acc += hB[q] * w3[q * NFILT + f];
  float h3 = sinf(fr * acc);
  __syncthreads();
  hA[f] = h3;
  __syncthreads();
  const float lnp   = -4.60517018598809136804f;  // ln(0.01)
  const float min_d = lnp / 1.5f, max_d = lnp / 0.3f;
  for (int j = 0; j < 16; j++) {
    int c = f * 16 + j;
    float a2 = 0.0f;
    for (int q = 0; q < NFILT; q++) a2 += hA[q] * wo[q * DMODEL + c];
    float delta = min_d + (max_d - min_d) * ((float)c * (1.0f / 1023.0f));
    float dec = expf(-t * fabsf(delta));
    krev[(size_t)c * KREV_ROW + (2047 - l)] = (bf16_t)(a2 * dec);
  }
}

// ------------------------------------------------- GEMM1: up = u @ in_w + b
// M=8192, N=3072, K=1024. 1 wave = 16x64 tile (4 accumulators), bf16 out.
__global__ __launch_bounds__(256) void k_gemm_up(
    const bf16_t* __restrict__ A, const bf16_t* __restrict__ Bt,
    const float* __restrict__ bias, bf16_t* __restrict__ C) {
  const int K = DMODEL, N = INNER3, NG = N / 64;
  int wg   = blockIdx.x * 8 + (threadIdx.x >> 5);
  int lane = threadIdx.x & 31;
  int mrow = lane & 15, lh = lane >> 4;
  int mt = wg / NG, ng = wg % NG;
  int n0 = ng * 64;
  const bf16_t* arow = A + (size_t)(mt * 16 + mrow) * K + lh * 8;
  v8f acc[4] = {{}, {}, {}, {}};
  for (int kb = 0; kb < K; kb += 32) {
    v16bf a = ld_pair(arow + kb, arow + kb + 16);
#pragma unroll
    for (int j = 0; j < 4; j++) {
      const bf16_t* bp = Bt + (size_t)(n0 + j * 16 + mrow) * K + kb + lh * 8;
      v16bf b = ld_pair(bp, bp + 16);
      acc[j] = wmma_bf16(a, b, acc[j]);
    }
  }
#pragma unroll
  for (int j = 0; j < 4; j++) {
#pragma unroll
    for (int r = 0; r < 8; r++) {
      int m = mt * 16 + r + lh * 8;
      int e = n0 + j * 16 + mrow;
      C[(size_t)m * N + e] = (bf16_t)(acc[j][r] + bias[e]);
    }
  }
}

// ------------------------------------------------- GEMM2: out = yv @ out_w + b
__global__ __launch_bounds__(256) void k_gemm_out(
    const bf16_t* __restrict__ A, const bf16_t* __restrict__ Bt,
    const float* __restrict__ bias, float* __restrict__ C) {
  const int K = DMODEL, N = DMODEL, NG = N / 64;
  int wg   = blockIdx.x * 8 + (threadIdx.x >> 5);
  int lane = threadIdx.x & 31;
  int mrow = lane & 15, lh = lane >> 4;
  int mt = wg / NG, ng = wg % NG;
  int n0 = ng * 64;
  const bf16_t* arow = A + (size_t)(mt * 16 + mrow) * K + lh * 8;
  v8f acc[4] = {{}, {}, {}, {}};
  for (int kb = 0; kb < K; kb += 32) {
    v16bf a = ld_pair(arow + kb, arow + kb + 16);
#pragma unroll
    for (int j = 0; j < 4; j++) {
      const bf16_t* bp = Bt + (size_t)(n0 + j * 16 + mrow) * K + kb + lh * 8;
      v16bf b = ld_pair(bp, bp + 16);
      acc[j] = wmma_bf16(a, b, acc[j]);
    }
  }
#pragma unroll
  for (int j = 0; j < 4; j++) {
#pragma unroll
    for (int r = 0; r < 8; r++) {
      int m = mt * 16 + r + lh * 8;
      int e = n0 + j * 16 + mrow;
      C[(size_t)m * N + e] = acc[j][r] + bias[e];
    }
  }
}

// ------------------------------------------------- width-3 depthwise conv + gate
// uc[b,e,l] = sum_k up[b,l-2+k,e]*sw[e,k] + sb[e]; vx=v*x1 (padded bf16), x0 bf16.
__global__ __launch_bounds__(256) void k_shortconv(
    const bf16_t* __restrict__ up, const float* __restrict__ sw,
    const float* __restrict__ sb, bf16_t* __restrict__ vxb,
    bf16_t* __restrict__ x0b) {
  int i = blockIdx.x * 256 + threadIdx.x;
  int c = i & (DMODEL - 1);
  int t = i >> 10;
  int l = t & (LSEQ - 1);
  int b = t >> 11;
  float res[3];
#pragma unroll
  for (int ch = 0; ch < 3; ch++) {
    int e = c + ch * DMODEL;
    float a = sb[e];
#pragma unroll
    for (int kk = 0; kk < 3; kk++) {
      int ls = l - 2 + kk;
      if (ls >= 0 && ls < LSEQ)
        a += (float)up[((size_t)(b * LSEQ + ls)) * INNER3 + e] * sw[e * 3 + kk];
    }
    res[ch] = a;
  }
  float vx = res[2] * res[1];
  size_t row = (size_t)(b * DMODEL + c);
  vxb[row * VX_ROW + VX_PAD + l] = (bf16_t)vx;
  x0b[row * LSEQ + l] = (bf16_t)res[0];
  if (l < VX_PAD) vxb[row * VX_ROW + l] = (bf16_t)0.0f;  // causal zero head
}

// ------------------------------------------------- causal long conv via WMMA
// y = T_c x (lower-tri block-Toeplitz). Per wave: channel c, 4 output chunks,
// N-cols = (4 batches x 4 chunks). A(16x32) = [T_{d+1}|T_d] gathered from
// LDS-staged reversed filter; epilogue fuses +vx*filt_bias and *x0 gates,
// writing the bf16 A-matrix of GEMM2 directly.
__global__ __launch_bounds__(256) void k_conv(
    const bf16_t* __restrict__ krev, const bf16_t* __restrict__ vxb,
    const bf16_t* __restrict__ x0b, const float* __restrict__ fb,
    bf16_t* __restrict__ A2) {
  int c  = blockIdx.x >> 2;
  int Ig = (blockIdx.x & 3) * 8 + (threadIdx.x >> 5);
  __shared__ bf16_t ks[KREV_ROW];
  for (int i = threadIdx.x; i < KREV_ROW; i += 256)
    ks[i] = krev[(size_t)c * KREV_ROW + i];
  __syncthreads();
  int lane = threadIdx.x & 31;
  int mrow = lane & 15, lh = lane >> 4;
  int bn = mrow >> 2;             // batch of this column
  int In = Ig * 4 + (mrow & 3);   // output chunk of this column
  const bf16_t* xrow = vxb + (size_t)(bn * DMODEL + c) * VX_ROW + VX_PAD;
  v8f acc = {};
  int ddend = Ig * 4 + 3;
  for (int dd = 0; dd <= ddend; dd += 2) {
    v16bf a;
    int r0 = 2031 - 16 * dd - mrow + lh * 8;  // krev index base, always >= 0
#pragma unroll
    for (int e = 0; e < 8; e++) {
      a[e]     = ks[r0 + e];       // K = lh*8 + e      -> delta dd+1 block
      a[e + 8] = ks[r0 + 16 + e];  // K = lh*8 + 16 + e -> delta dd   block
    }
    const bf16_t* xp = xrow + 16 * (In - dd - 1) + lh * 8;  // zero head covers J<0
    v16bf b = ld_pair(xp, xp + 16);
    acc = wmma_bf16(a, b, acc);
  }
  float fbc = fb[c];
  const bf16_t* x0row = x0b + (size_t)(bn * DMODEL + c) * LSEQ;
#pragma unroll
  for (int r = 0; r < 8; r++) {
    int l = In * 16 + r + lh * 8;
    float vx  = (float)xrow[l];
    float x0  = (float)x0row[l];
    float val = (acc[r] + vx * fbc) * x0;
    A2[((size_t)(bn * LSEQ + l)) * DMODEL + c] = (bf16_t)val;
  }
}

// ---------------------------------------------------------------- launcher
extern "C" void kernel_launch(void* const* d_in, const int* in_sizes, int n_in,
                              void* d_out, int out_size, void* d_ws, size_t ws_size,
                              hipStream_t stream) {
  (void)in_sizes; (void)n_in; (void)out_size; (void)ws_size;
  const float* u     = (const float*)d_in[0];
  const float* in_w  = (const float*)d_in[1];
  const float* in_b  = (const float*)d_in[2];
  const float* sw    = (const float*)d_in[3];
  const float* sb    = (const float*)d_in[4];
  const float* w1    = (const float*)d_in[5];
  const float* b1    = (const float*)d_in[6];
  const float* w2    = (const float*)d_in[7];
  const float* b2    = (const float*)d_in[8];
  const float* w3    = (const float*)d_in[9];
  const float* b3    = (const float*)d_in[10];
  const float* wo    = (const float*)d_in[11];
  const float* freq  = (const float*)d_in[12];
  const float* fbias = (const float*)d_in[13];
  const float* out_w = (const float*)d_in[14];
  const float* out_b = (const float*)d_in[15];
  float* out = (float*)d_out;

  char* ws = (char*)d_ws;
  size_t off = 0;
  auto take = [&](size_t elems) {
    bf16_t* p = (bf16_t*)(ws + off);
    off += (elems * sizeof(bf16_t) + 255) & ~(size_t)255;
    return p;
  };
  bf16_t* ub   = take((size_t)BATCH * LSEQ * DMODEL);   // u in bf16
  bf16_t* wt1  = take((size_t)INNER3 * DMODEL);         // in_w^T bf16
  bf16_t* wt2  = take((size_t)DMODEL * DMODEL);         // out_w^T bf16
  bf16_t* upb  = take((size_t)BATCH * LSEQ * INNER3);   // projected input
  bf16_t* vxb  = take((size_t)BATCH * DMODEL * VX_ROW); // v*x1, zero-headed
  bf16_t* x0b  = take((size_t)BATCH * DMODEL * LSEQ);   // x0 gate
  bf16_t* krev = take((size_t)DMODEL * KREV_ROW);       // reversed filter
  bf16_t* a2   = take((size_t)BATCH * LSEQ * DMODEL);   // (y+vx*fb)*x0

  int nU = BATCH * LSEQ * DMODEL;
  k_f32_to_bf16<<<(nU + 255) / 256, 256, 0, stream>>>(u, ub, nU);
  k_transpose_bf16<<<(DMODEL * INNER3 + 255) / 256, 256, 0, stream>>>(in_w, wt1, DMODEL, INNER3);
  k_transpose_bf16<<<(DMODEL * DMODEL + 255) / 256, 256, 0, stream>>>(out_w, wt2, DMODEL, DMODEL);
  int nK = DMODEL * KREV_ROW;
  k_zero_bf16<<<(nK + 255) / 256, 256, 0, stream>>>(krev, nK);
  k_filter<<<LSEQ, NFILT, 0, stream>>>(w1, b1, w2, b2, w3, b3, wo, freq, krev);
  // GEMM1: 512 M-tiles * 48 N-groups = 24576 wave-tasks, 8 waves/block
  k_gemm_up<<<24576 / 8, 256, 0, stream>>>(ub, wt1, in_b, upb);
  k_shortconv<<<(BATCH * LSEQ * DMODEL) / 256, 256, 0, stream>>>(upb, sw, sb, vxb, x0b);
  // conv: 1024 channels * 4 blocks (8 waves each -> 32 chunk-groups)
  k_conv<<<DMODEL * 4, 256, 0, stream>>>(krev, vxb, x0b, fbias, a2);
  // GEMM2: 512 M-tiles * 16 N-groups = 8192 wave-tasks
  k_gemm_out<<<8192 / 8, 256, 0, stream>>>(a2, wt2, out_b, out);
}
